// SingleAttentionLayer_75866302316655
// MI455X (gfx1250) — matrix operand order repaired and checked by reading, not compiled, not run on hardware
//
#include <hip/hip_runtime.h>
#include <hip/hip_bf16.h>

// MI455X / gfx1250, wave32. WMMA f16->f32 GEMMs + async global->LDS staging.

typedef __attribute__((ext_vector_type(16))) _Float16 v16h_t;
typedef __attribute__((ext_vector_type(8)))  _Float16 v8h_t;
typedef __attribute__((ext_vector_type(8)))  float    v8f_t;
typedef __attribute__((ext_vector_type(4)))  float    v4f_t;

static constexpr int Bn   = 8;
static constexpr int Lseq = 2048;
static constexpr int Dm   = 1024;
static constexpr int NTOK = Bn * Lseq;   // 16384 rows

// ---- gfx1250 async copy: global -> LDS, 16 bytes per lane (ASYNCcnt) ----
__device__ __forceinline__ void async_b128(const void* sbase, unsigned gbyte, unsigned ldsbyte) {
  asm volatile("global_load_async_to_lds_b128 %0, %1, %2"
               :: "v"(ldsbyte), "v"(gbyte), "s"(sbase)
               : "memory");
}
__device__ __forceinline__ void wait_async0() {
#if __has_builtin(__builtin_amdgcn_s_wait_asynccnt)
  __builtin_amdgcn_s_wait_asynccnt(0);
#else
  asm volatile("s_wait_asynccnt 0x0" ::: "memory");
#endif
}
__device__ __forceinline__ unsigned lds_off(const void* p) {
  return (unsigned)(size_t)p;   // low 32 bits of flat-shared address = LDS offset
}

// ------------------------------------------------------------------
// Row-zero flags (padding masks): flag[row] = (sum_d x[row,d] == 0)
// ------------------------------------------------------------------
__global__ __launch_bounds__(256)
void rowmask_kernel(const float* __restrict__ x, int* __restrict__ flag, int cols) {
  const int w = threadIdx.x >> 5, lane = threadIdx.x & 31;
  const int row = blockIdx.x * 8 + w;
  const float* p = x + (size_t)row * cols;
  float s = 0.0f;
  for (int i = lane; i < cols; i += 32) s += p[i];
#pragma unroll
  for (int off = 16; off; off >>= 1) s += __shfl_xor(s, off, 32);
  if (lane == 0) flag[row] = (s == 0.0f) ? 1 : 0;
}

// ------------------------------------------------------------------
// Generic WMMA GEMM. Block tile 128x128, K-step 32, 8 waves.
// Wave -> 32(M) x 64(N) strip: acc[2][4], 8x v_wmma_f32_16x16x32_f16 / step.
//   A: f32 (convert+ds_store) or f16 (async global->LDS), row-major, lda
//   B: !BTRANS: LDSB[n][k] = Bsrc[n*ldb + k]   (f32 convert, or f16 async)
//      BTRANS : LDSB[n][k] = Bsrc[k*ldb + n]   (f16 scatter staging)
// ------------------------------------------------------------------
template<bool AF16, bool BF16, bool BTRANS, bool HASBIAS, bool HASRES, bool OUT16, bool OUT32>
__global__ __launch_bounds__(256)
void gemm_wmma_kernel(const void* __restrict__ Aptr, long batchA, int lda,
                      const void* __restrict__ Bptr, long batchB, int ldb,
                      const float* __restrict__ bias,
                      const float* __restrict__ res, long batchR,
                      _Float16* __restrict__ out16, long batchO16,
                      float* __restrict__ out32, long batchO32,
                      int ldo, int K, float scale)
{
  constexpr int BM = 128, BN = 128, BK = 32, LDA = 40, LDB = 40;
  __shared__ _Float16 lA[BM * LDA];
  __shared__ _Float16 lB[BN * LDB];

  const int z  = blockIdx.z;
  const int m0 = blockIdx.x * BM;
  const int n0 = blockIdx.y * BN;
  const int t  = threadIdx.x;
  const int w  = t >> 5, lane = t & 31;
  const int wm = w & 3, wn = w >> 2;        // 4x2 wave grid over 128x128

  const _Float16* A16 = (const _Float16*)Aptr + (size_t)z * batchA;
  const float*    A32 = (const float*)Aptr    + (size_t)z * batchA;
  const _Float16* B16 = (const _Float16*)Bptr + (size_t)z * batchB;
  const float*    B32 = (const float*)Bptr    + (size_t)z * batchB;

  v8f_t acc[2][4];
#pragma unroll
  for (int i = 0; i < 2; ++i)
#pragma unroll
    for (int j = 0; j < 4; ++j) acc[i][j] = (v8f_t){0,0,0,0,0,0,0,0};

  const int sRow  = t >> 1, sHalf = t & 1;  // 128 rows x 2 half-rows of 16 elems
  const int btK   = t >> 3, btG   = t & 7;  // BTRANS: 32 k-rows x 8 groups of 16 n

  for (int k0 = 0; k0 < K; k0 += BK) {
    bool used_async = false;

    // ---------------- stage A tile (128 x 32) ----------------
    {
      _Float16* dst = lA + sRow * LDA + sHalf * 16;
      if constexpr (AF16) {
        const unsigned gb = (unsigned)(((size_t)(m0 + sRow) * lda + k0 + sHalf * 16) * 2);
        const unsigned lb = lds_off(dst);
        async_b128(A16, gb, lb);
        async_b128(A16, gb + 16u, lb + 16u);
        used_async = true;
      } else {
        const v4f_t* s = (const v4f_t*)(A32 + (size_t)(m0 + sRow) * lda + k0 + sHalf * 16);
#pragma unroll
        for (int u = 0; u < 4; ++u) {
          v4f_t f = s[u];
          dst[u*4+0] = (_Float16)f.x; dst[u*4+1] = (_Float16)f.y;
          dst[u*4+2] = (_Float16)f.z; dst[u*4+3] = (_Float16)f.w;
        }
      }
    }
    // ---------------- stage B tile (128 n x 32 k) -> LDS [n][k] ----------------
    if constexpr (!BTRANS) {
      _Float16* dst = lB + sRow * LDB + sHalf * 16;
      if constexpr (BF16) {
        const unsigned gb = (unsigned)(((size_t)(n0 + sRow) * ldb + k0 + sHalf * 16) * 2);
        const unsigned lb = lds_off(dst);
        async_b128(B16, gb, lb);
        async_b128(B16, gb + 16u, lb + 16u);
        used_async = true;
      } else {
        const v4f_t* s = (const v4f_t*)(B32 + (size_t)(n0 + sRow) * ldb + k0 + sHalf * 16);
#pragma unroll
        for (int u = 0; u < 4; ++u) {
          v4f_t f = s[u];
          dst[u*4+0] = (_Float16)f.x; dst[u*4+1] = (_Float16)f.y;
          dst[u*4+2] = (_Float16)f.z; dst[u*4+3] = (_Float16)f.w;
        }
      }
    } else {
      const _Float16* src = B16 + (size_t)(k0 + btK) * ldb + n0 + btG * 16;
      const v8h_t x0 = *(const v8h_t*)src;
      const v8h_t x1 = *(const v8h_t*)(src + 8);
#pragma unroll
      for (int u = 0; u < 8; ++u) {
        lB[(btG * 16 + u) * LDB + btK]     = x0[u];
        lB[(btG * 16 + 8 + u) * LDB + btK] = x1[u];
      }
    }
    if (used_async) wait_async0();
    __syncthreads();

    // ---- A fragments: lane = M + 16*kh; runs at kh*8 and kh*8+16 ----
    v16h_t afrag[2];
#pragma unroll
    for (int mt = 0; mt < 2; ++mt) {
      const _Float16* ap = lA + (wm * 32 + mt * 16 + (lane & 15)) * LDA + (lane >> 4) * 8;
      const v8h_t a0 = *(const v8h_t*)ap;
      const v8h_t a1 = *(const v8h_t*)(ap + 16);
#pragma unroll
      for (int u = 0; u < 8; ++u) { afrag[mt][u] = a0[u]; afrag[mt][u + 8] = a1[u]; }
    }
#pragma unroll
    for (int nt = 0; nt < 4; ++nt) {
      // B fragment: lane = N + 16*kh; runs at kh*8 and kh*8+16
      const _Float16* bp = lB + (wn * 64 + nt * 16 + (lane & 15)) * LDB + (lane >> 4) * 8;
      const v8h_t b0 = *(const v8h_t*)bp;
      const v8h_t b1 = *(const v8h_t*)(bp + 16);
      v16h_t bfrag;
#pragma unroll
      for (int u = 0; u < 8; ++u) { bfrag[u] = b0[u]; bfrag[u + 8] = b1[u]; }
#pragma unroll
      for (int mt = 0; mt < 2; ++mt)
        acc[mt][nt] = __builtin_amdgcn_wmma_f32_16x16x32_f16(
            false, afrag[mt], false, bfrag, (short)0, acc[mt][nt], false, false);
    }
    __syncthreads();
  }

  // ---- epilogue: D layout VGPR r -> M = r + 8*(lane>=16), N = lane&15 ----
#pragma unroll
  for (int mt = 0; mt < 2; ++mt) {
    const int rowbase = m0 + wm * 32 + mt * 16 + (lane >> 4) * 8;
#pragma unroll
    for (int nt = 0; nt < 4; ++nt) {
      const int col = n0 + wn * 64 + nt * 16 + (lane & 15);
      const float bb = HASBIAS ? bias[col] : 0.0f;
#pragma unroll
      for (int r = 0; r < 8; ++r) {
        const int row = rowbase + r;
        float val = acc[mt][nt][r] * scale + bb;
        if constexpr (HASRES) val += res[(size_t)z * batchR + (size_t)row * ldo + col];
        if constexpr (OUT16)
          out16[(size_t)z * batchO16 + (size_t)row * ldo + col] = (_Float16)val;
        if constexpr (OUT32)
          out32[(size_t)z * batchO32 + (size_t)row * ldo + col] = val;
      }
    }
  }
}

// ------------------------------------------------------------------
// Masked softmax over S rows -> P (f16).
// ------------------------------------------------------------------
__global__ __launch_bounds__(256)
void softmax_mask_kernel(const float* __restrict__ S, _Float16* __restrict__ P,
                         const int* __restrict__ qflag, const int* __restrict__ kflag)
{
  const int b = blockIdx.y, i = blockIdx.x, t = threadIdx.x;
  const size_t rowoff = ((size_t)b * Lseq + i) * (size_t)Lseq;
  _Float16* prow = P + rowoff;
  if (qflag[b * Lseq + i]) {
    for (int j = t; j < Lseq; j += 256) prow[j] = (_Float16)0.0f;
    return;
  }
  const float* srow = S + rowoff;
  const int* kf = kflag + b * Lseq;

  float v[8];
  float mx = -3.4e38f;
#pragma unroll
  for (int u = 0; u < 8; ++u) {
    const int j = t + u * 256;
    const float x = kf[j] ? -__builtin_inff() : srow[j];
    v[u] = x; mx = fmaxf(mx, x);
  }
  __shared__ float red[8];
  const int w = t >> 5, lane = t & 31;
#pragma unroll
  for (int off = 16; off; off >>= 1) mx = fmaxf(mx, __shfl_xor(mx, off, 32));
  if (lane == 0) red[w] = mx;
  __syncthreads();
#pragma unroll
  for (int u = 0; u < 8; ++u) mx = fmaxf(mx, red[u]);

  float ssum = 0.0f;
#pragma unroll
  for (int u = 0; u < 8; ++u) { const float e = __expf(v[u] - mx); v[u] = e; ssum += e; }
#pragma unroll
  for (int off = 16; off; off >>= 1) ssum += __shfl_xor(ssum, off, 32);
  __syncthreads();
  if (lane == 0) red[w] = ssum;
  __syncthreads();
  float tot = 0.0f;
#pragma unroll
  for (int u = 0; u < 8; ++u) tot += red[u];
  const float inv = 1.0f / tot;
#pragma unroll
  for (int u = 0; u < 8; ++u) prow[t + u * 256] = (_Float16)(v[u] * inv);
}

// ------------------------------------------------------------------
// In-place LayerNorm (no affine), eps=1e-5. One block per row of 1024.
// ------------------------------------------------------------------
__global__ __launch_bounds__(256)
void layernorm_kernel(float* __restrict__ X)
{
  const int t = threadIdx.x;
  float* row = X + (size_t)blockIdx.x * Dm;
  float v[4];
  float s = 0.0f;
#pragma unroll
  for (int u = 0; u < 4; ++u) { v[u] = row[t + u * 256]; s += v[u]; }

  __shared__ float red[8];
  const int w = t >> 5, lane = t & 31;
#pragma unroll
  for (int off = 16; off; off >>= 1) s += __shfl_xor(s, off, 32);
  if (lane == 0) red[w] = s;
  __syncthreads();
  float mean = 0.0f;
#pragma unroll
  for (int u = 0; u < 8; ++u) mean += red[u];
  mean *= (1.0f / Dm);
  __syncthreads();

  float q = 0.0f;
#pragma unroll
  for (int u = 0; u < 4; ++u) { const float d = v[u] - mean; q += d * d; }
#pragma unroll
  for (int off = 16; off; off >>= 1) q += __shfl_xor(q, off, 32);
  if (lane == 0) red[w] = q;
  __syncthreads();
  float var = 0.0f;
#pragma unroll
  for (int u = 0; u < 8; ++u) var += red[u];
  var *= (1.0f / Dm);
  const float inv = rsqrtf(var + 1e-5f);
#pragma unroll
  for (int u = 0; u < 4; ++u) row[t + u * 256] = (v[u] - mean) * inv;
}

// ------------------------------------------------------------------
extern "C" void kernel_launch(void* const* d_in, const int* in_sizes, int n_in,
                              void* d_out, int out_size, void* d_ws, size_t ws_size,
                              hipStream_t stream)
{
  (void)in_sizes; (void)n_in; (void)out_size; (void)ws_size;
  const float* q  = (const float*)d_in[0];
  const float* k  = (const float*)d_in[1];
  const float* v  = (const float*)d_in[2];
  const float* Wq = (const float*)d_in[3];
  const float* bq = (const float*)d_in[4];
  const float* Wk = (const float*)d_in[5];
  const float* bk = (const float*)d_in[6];
  const float* Wv = (const float*)d_in[7];
  const float* bv = (const float*)d_in[8];
  const float* Wo = (const float*)d_in[9];
  const float* bo = (const float*)d_in[10];
  float* out = (float*)d_out;

  char* ws = (char*)d_ws;
  size_t off = 0;
  auto wsalloc = [&](size_t bytes) -> void* {
    void* p = ws + off;
    off += (bytes + 255) & ~(size_t)255;
    return p;
  };
  _Float16* qh  = (_Float16*)wsalloc((size_t)NTOK * Dm * 2);
  _Float16* kh  = (_Float16*)wsalloc((size_t)NTOK * Dm * 2);
  _Float16* vh  = (_Float16*)wsalloc((size_t)NTOK * Dm * 2);
  float*    qpf = (float*)   wsalloc((size_t)NTOK * Dm * 4);
  float*    S   = (float*)   wsalloc((size_t)Bn * Lseq * Lseq * 4);
  _Float16* P   = (_Float16*)wsalloc((size_t)Bn * Lseq * Lseq * 2);
  _Float16* Oh  = (_Float16*)wsalloc((size_t)NTOK * Dm * 2);
  int*      qflag = (int*)   wsalloc((size_t)NTOK * 4);
  int*      kflag = (int*)   wsalloc((size_t)NTOK * 4);

  // 1) padding masks from raw q / k
  rowmask_kernel<<<NTOK / 8, 256, 0, stream>>>(q, qflag, Dm);
  rowmask_kernel<<<NTOK / 8, 256, 0, stream>>>(k, kflag, Dm);

  // 2) projections: y = x W^T + b  (f16 copies for WMMA; qp also f32 for residual)
  dim3 gproj(NTOK / 128, Dm / 128, 1);
  gemm_wmma_kernel<false,false,false,true,false,true,true><<<gproj, 256, 0, stream>>>(
      q, 0, Dm, Wq, 0, Dm, bq, nullptr, 0, qh, 0, qpf, 0, Dm, Dm, 1.0f);
  gemm_wmma_kernel<false,false,false,true,false,true,false><<<gproj, 256, 0, stream>>>(
      k, 0, Dm, Wk, 0, Dm, bk, nullptr, 0, kh, 0, nullptr, 0, Dm, Dm, 1.0f);
  gemm_wmma_kernel<false,false,false,true,false,true,false><<<gproj, 256, 0, stream>>>(
      v, 0, Dm, Wv, 0, Dm, bv, nullptr, 0, vh, 0, nullptr, 0, Dm, Dm, 1.0f);

  // 3) S = (qp kp^T) * D^-1/2, batched over B  (both operands async-staged f16)
  dim3 gs(Lseq / 128, Lseq / 128, Bn);
  gemm_wmma_kernel<true,true,false,false,false,false,true><<<gs, 256, 0, stream>>>(
      qh, (long)Lseq * Dm, Dm, kh, (long)Lseq * Dm, Dm, nullptr, nullptr, 0,
      nullptr, 0, S, (long)Lseq * Lseq, Lseq, Dm, 0.03125f);

  // 4) masked softmax -> P (f16)
  dim3 gsm(Lseq, Bn, 1);
  softmax_mask_kernel<<<gsm, 256, 0, stream>>>(S, P, qflag, kflag);

  // 5) O = P vp  (A async-staged, B transpose-staged)
  dim3 go(Lseq / 128, Dm / 128, Bn);
  gemm_wmma_kernel<true,true,true,false,false,true,false><<<go, 256, 0, stream>>>(
      P, (long)Lseq * Lseq, Lseq, vh, (long)Lseq * Dm, Dm, nullptr, nullptr, 0,
      Oh, (long)Lseq * Dm, nullptr, 0, Dm, Lseq, 1.0f);

  // 6) Y = O Wo^T + bo + qp  -> d_out (f32), then in-place LayerNorm
  dim3 gy(NTOK / 128, Dm / 128, 1);
  gemm_wmma_kernel<true,false,false,true,true,false,true><<<gy, 256, 0, stream>>>(
      Oh, 0, Dm, Wo, 0, Dm, bo, qpf, 0, nullptr, 0, out, 0, Dm, Dm, 1.0f);

  layernorm_kernel<<<NTOK, 256, 0, stream>>>(out);
}
